// DecoderMultiHeadAttention_49271864819774
// MI455X (gfx1250) — matrix operand order repaired and checked by reading, not compiled, not run on hardware
//
#include <hip/hip_runtime.h>
#include <hip/hip_bf16.h>

// ---------------------------------------------------------------------------
// DecoderMultiHeadAttention for MI455X (gfx1250), wave32 + WMMA bf16.
// B=8, S=1024, E=768, H=12, D=64. All GEMMs use v_wmma_f32_16x16x32_bf16.
// K-tiles in the attention kernel use GLOBAL_LOAD_ASYNC_TO_LDS (ASYNCcnt).
// ---------------------------------------------------------------------------

constexpr int NB = 8;
constexpr int NS = 1024;
constexpr int NE = 768;
constexpr int NH = 12;
constexpr int ND = 64;
#define ATT_SCALE 0.125f   // 1/sqrt(64)

typedef __bf16 bf16;
typedef __attribute__((ext_vector_type(16))) __bf16 v16bf;
typedef __attribute__((ext_vector_type(8)))  __bf16 v8bf;
typedef __attribute__((ext_vector_type(8)))  float  v8f;
typedef __attribute__((ext_vector_type(4)))  int    v4i;

#define AS_GLOBAL __attribute__((address_space(1)))
#define AS_LDS    __attribute__((address_space(3)))

// Async global->LDS path (CDNA5): only in the device pass, only if the
// toolchain declares both the copy builtin and the asynccnt wait.
#if defined(__gfx1250__) && defined(__AMDGCN__)
#if __has_builtin(__builtin_amdgcn_global_load_async_to_lds_b128) && \
    __has_builtin(__builtin_amdgcn_s_wait_asynccnt)
#define USE_ASYNC_LDS 1
#endif
#endif

// fp32 -> bf16 via native fptrunc (backend selects the hardware convert).
__device__ __forceinline__ bf16 f2bf(float f) { return (bf16)f; }

// Pack two fp32 into a dword of two bf16 (lo = a, hi = b).
__device__ __forceinline__ unsigned pack2(float a, float b) {
  bf16 l = (bf16)a, h = (bf16)b;
  unsigned short ul, uh;
  __builtin_memcpy(&ul, &l, 2);
  __builtin_memcpy(&uh, &h, 2);
  return (unsigned)ul | ((unsigned)uh << 16);
}

__device__ __forceinline__ v16bf ld16(const bf16* p0, const bf16* p1) {
  union { v16bf v; v8bf h[2]; } u;
  u.h[0] = *(const v8bf*)p0;   // 16B load
  u.h[1] = *(const v8bf*)p1;   // 16B load
  return u.v;
}

// A-fragment (16x32 bf16): `row` points at this lane's row start + k-window base.
// lanes 0-15: K = 0..7 and 16..23 ; lanes 16-31: K = 8..15 and 24..31.
__device__ __forceinline__ v16bf frag_a(const bf16* row, int hi) {
  return ld16(row + 8 * hi, row + 16 + 8 * hi);
}

// B-fragment (32x16 bf16) from an N-major ("transposed") LDS tile:
// `nrow` points at this lane's column (N = lane%16) + k-window base.
// lanes 0-15: K = 0..7 and 16..23 ; lanes 16-31: K = 8..15 and 24..31.
__device__ __forceinline__ v16bf frag_b(const bf16* nrow, int hi) {
  return ld16(nrow + 8 * hi, nrow + 16 + 8 * hi);
}

__device__ __forceinline__ v8f wmma_bf16(v16bf a, v16bf b, v8f c) {
  return __builtin_amdgcn_wmma_f32_16x16x32_bf16(
      /*neg_a=*/false, a, /*neg_b=*/false, b,
      /*c_mod=*/(short)0, c, /*reuse_a=*/false, /*reuse_b=*/false);
}

// ---------------------------------------------------------------------------
// Kernel 1: per-head projection  Y[b,h,s,d] = X[b,s,:] . W[h,:,d] + bias[h,d]
// grid = (NS/64, NB*NH), block = 256 (8 waves). 64x64 tile, K-step 32.
// ---------------------------------------------------------------------------
__global__ __launch_bounds__(256)
void qkv_proj_kernel(const float* __restrict__ X, const float* __restrict__ W,
                     const float* __restrict__ bias, bf16* __restrict__ Y) {
  __shared__ __align__(16) bf16 As[64 * 32];   // [m][k], row stride 32
  __shared__ __align__(16) bf16 Bt[64 * 32];   // [n][k] transposed W tile

  const int tid  = threadIdx.x;
  const int lane = tid & 31;
  const int wave = tid >> 5;
  const int lrow = lane & 15;
  const int hi   = (lane >> 4) & 1;

  const int bh = blockIdx.y;
  const int h  = bh % NH;
  const int b  = bh / NH;
  const int m0 = blockIdx.x * 64;
  const int msub = (wave & 3) * 16;
  const int nsub = (wave >> 2) * 32;

  const float* Xb = X + (size_t)b * NS * NE;
  const float* Wh = W + (size_t)h * NE * ND;

  v8f acc0 = {}, acc1 = {};

  for (int e0 = 0; e0 < NE; e0 += 32) {
    // A tile: 64x32 fp32 -> bf16, packed-pair dword writes
    #pragma unroll
    for (int i = 0; i < 4; ++i) {
      int p = tid + i * 256;             // pair id 0..1023
      int r = p >> 4, cp = (p & 15) * 2;
      float2 xv = *(const float2*)(Xb + (size_t)(m0 + r) * NE + e0 + cp);
      ((unsigned*)As)[r * 16 + (p & 15)] = pack2(xv.x, xv.y);
    }
    // B tile transposed: Bt[n][k] = W[h][e0+k][n], packed pairs along k
    #pragma unroll
    for (int i = 0; i < 4; ++i) {
      int p = tid + i * 256;
      int n = p >> 4, kp = (p & 15) * 2;
      float w0 = Wh[(size_t)(e0 + kp) * ND + n];
      float w1 = Wh[(size_t)(e0 + kp + 1) * ND + n];
      ((unsigned*)Bt)[n * 16 + (p & 15)] = pack2(w0, w1);
    }
    __syncthreads();

    v16bf a  = frag_a(&As[(msub + lrow) * 32], hi);
    v16bf b0 = frag_b(&Bt[(nsub + lrow) * 32], hi);
    v16bf b1 = frag_b(&Bt[(nsub + 16 + lrow) * 32], hi);
    acc0 = wmma_bf16(a, b0, acc0);
    acc1 = wmma_bf16(a, b1, acc1);
    __syncthreads();
  }

  // Epilogue: bias add, store bf16. C layout: lane holds (M=8*hi+r, N=lane%16).
  bf16* Yh = Y + (size_t)bh * NS * ND;
  #pragma unroll
  for (int r = 0; r < 8; ++r) {
    int row = m0 + msub + hi * 8 + r;
    int c0  = nsub + lrow;
    int c1  = c0 + 16;
    Yh[(size_t)row * ND + c0] = f2bf(acc0[r] + bias[h * ND + c0]);
    Yh[(size_t)row * ND + c1] = f2bf(acc1[r] + bias[h * ND + c1]);
  }
}

// ---------------------------------------------------------------------------
// Kernel 2: causal flash attention per (b,h).
// grid = (NS/128, NB*NH), block = 256. Each wave owns 16 complete query rows,
// so softmax reductions are 16-lane half-wave shuffles. Key/value tiles of 64.
// ---------------------------------------------------------------------------
__global__ __launch_bounds__(256)
void attn_kernel(const bf16* __restrict__ Qp, const bf16* __restrict__ Kp,
                 const bf16* __restrict__ Vp, bf16* __restrict__ Hd) {
  __shared__ __align__(16) bf16 Kt[64 * 64];        // [t][d]  (== Bt for Q.K^T)
  __shared__ __align__(16) bf16 Vt[64 * 64];        // [d][t]  (Bt for P.V)
  __shared__ __align__(16) bf16 Pw[8 * 16 * 64];    // per-wave P, row-major 16x64

  const int tid  = threadIdx.x;
  const int lane = tid & 31;
  const int wave = tid >> 5;
  const int lrow = lane & 15;
  const int hi   = (lane >> 4) & 1;

  const int bh = blockIdx.y;
  const int h  = bh % NH;
  const int b  = bh / NH;
  const int m0 = blockIdx.x * 128;
  const int mw = m0 + wave * 16;                    // this wave's first query row

  const bf16* Qh = Qp + (size_t)bh * NS * ND;
  const bf16* Kh = Kp + (size_t)bh * NS * ND;
  const bf16* Vh = Vp + (size_t)bh * NS * ND;
  const unsigned short* Vs = (const unsigned short*)Vh;

  // Q fragments for both 32-wide k-steps of D=64 (A-layout, straight from global)
  v16bf qf0, qf1;
  {
    const bf16* qrow = Qh + (size_t)(mw + lrow) * ND;
    qf0 = frag_a(qrow + 0,  hi);
    qf1 = frag_a(qrow + 32, hi);
  }

  float mrun[8], lrun[8];
  v8f oacc[4] = {};
  #pragma unroll
  for (int r = 0; r < 8; ++r) { mrun[r] = -1e30f; lrun[r] = 0.0f; }

  for (int t0 = 0; t0 < m0 + 128; t0 += 64) {
    // ---- K tile: straight global->LDS copy (async on CDNA5, ASYNCcnt) ----
#ifdef USE_ASYNC_LDS
    {
      const bf16* src = Kh + (size_t)t0 * ND;
      #pragma unroll
      for (int i = 0; i < 2; ++i) {
        int c = tid + i * 256;                     // 16B chunk id, 0..511
        __builtin_amdgcn_global_load_async_to_lds_b128(
            (AS_GLOBAL v4i*)(src + c * 8),
            (AS_LDS v4i*)(&Kt[c * 8]),
            /*offset=*/0, /*cpol=*/0);
      }
    }
#else
    {
      const uint4* src = (const uint4*)(Kh + (size_t)t0 * ND);
      #pragma unroll
      for (int i = 0; i < 2; ++i) {
        int c = tid + i * 256;
        ((uint4*)Kt)[c] = src[c];
      }
    }
#endif
    // ---- V tile transposed: Vt[d][t], packed t-pairs per dword ----
    #pragma unroll
    for (int i = 0; i < 8; ++i) {
      int p = tid + i * 256;                       // pair id 0..2047
      int d = p & 63, tp = (p >> 6) * 2;
      unsigned lo = Vs[(size_t)(t0 + tp) * ND + d];
      unsigned hi16 = Vs[(size_t)(t0 + tp + 1) * ND + d];
      ((unsigned*)Vt)[d * 32 + (p >> 6)] = lo | (hi16 << 16);
    }
    if (t0 + 64 < m0 + 128)                        // prefetch next K tile
      __builtin_prefetch(Kh + (size_t)(t0 + 64) * ND, 0, 0);
#ifdef USE_ASYNC_LDS
    __builtin_amdgcn_s_wait_asynccnt(0);           // own wave's async copies done
#endif
    __syncthreads();                               // all waves' tiles visible

    const bool active = (t0 <= mw + 15);           // wave-uniform
    if (active) {
      // scores = Q . K^T   (16x64 per wave)
      v8f s[4] = {};
      #pragma unroll
      for (int j = 0; j < 4; ++j) {
        s[j] = wmma_bf16(qf0, frag_b(&Kt[(j * 16 + lrow) * 64 + 0],  hi), s[j]);
        s[j] = wmma_bf16(qf1, frag_b(&Kt[(j * 16 + lrow) * 64 + 32], hi), s[j]);
      }

      // scale + causal mask + running row max
      float pm[4][8];
      float rmax[8];
      #pragma unroll
      for (int r = 0; r < 8; ++r) {
        const int row = mw + hi * 8 + r;
        float best = -1e30f;
        #pragma unroll
        for (int j = 0; j < 4; ++j) {
          const int col = t0 + j * 16 + lrow;
          float v = s[j][r] * ATT_SCALE;
          if (col > row) v = -1e30f;
          pm[j][r] = v;
          best = fmaxf(best, v);
        }
        rmax[r] = best;
      }
      #pragma unroll
      for (int off = 1; off < 16; off <<= 1)
        #pragma unroll
        for (int r = 0; r < 8; ++r)
          rmax[r] = fmaxf(rmax[r], __shfl_xor(rmax[r], off, 32));

      float alpha[8], rsum[8];
      #pragma unroll
      for (int r = 0; r < 8; ++r) {
        const float mnew = fmaxf(mrun[r], rmax[r]);
        alpha[r] = __expf(mrun[r] - mnew);
        mrun[r]  = mnew;
        float su = 0.0f;
        #pragma unroll
        for (int j = 0; j < 4; ++j) {
          const float p = __expf(pm[j][r] - mnew);
          pm[j][r] = p;
          su += p;
        }
        rsum[r] = su;
      }
      #pragma unroll
      for (int off = 1; off < 16; off <<= 1)
        #pragma unroll
        for (int r = 0; r < 8; ++r)
          rsum[r] += __shfl_xor(rsum[r], off, 32);
      #pragma unroll
      for (int r = 0; r < 8; ++r)
        lrun[r] = lrun[r] * alpha[r] + rsum[r];

      // rescale running O
      #pragma unroll
      for (int j = 0; j < 4; ++j)
        #pragma unroll
        for (int r = 0; r < 8; ++r)
          oacc[j][r] *= alpha[r];

      // C-layout -> A-layout reshape of P through per-wave LDS (wave-local,
      // DS ops are in-order within a wave; no block barrier required)
      bf16* P = &Pw[wave * 16 * 64];
      #pragma unroll
      for (int j = 0; j < 4; ++j)
        #pragma unroll
        for (int r = 0; r < 8; ++r)
          P[(hi * 8 + r) * 64 + (j * 16 + lrow)] = f2bf(pm[j][r]);

      // O += P . V
      #pragma unroll
      for (int ks = 0; ks < 2; ++ks) {
        v16bf af = frag_a(&P[lrow * 64 + ks * 32], hi);
        #pragma unroll
        for (int j = 0; j < 4; ++j)
          oacc[j] = wmma_bf16(af, frag_b(&Vt[(j * 16 + lrow) * 64 + ks * 32], hi), oacc[j]);
      }
    }
    __syncthreads();   // uniform: protects Kt/Vt reuse next iteration
  }

  // normalize and store heads concat [B,S,H*D] in bf16 for the final GEMM
  bf16* out = Hd + (size_t)b * NS * (NH * ND);
  #pragma unroll
  for (int j = 0; j < 4; ++j)
    #pragma unroll
    for (int r = 0; r < 8; ++r) {
      const int row = mw + hi * 8 + r;
      const int col = h * ND + j * 16 + lrow;
      out[(size_t)row * (NH * ND) + col] = f2bf(oacc[j][r] / lrun[r]);
    }
}

// ---------------------------------------------------------------------------
// Kernel 3: output projection  Out = heads(bf16) . Wo + bo   (fp32 out)
// grid = (B*S/64, E/64), block = 256. A-fragments straight from global.
// ---------------------------------------------------------------------------
__global__ __launch_bounds__(256)
void out_proj_kernel(const bf16* __restrict__ Hd, const float* __restrict__ Wo,
                     const float* __restrict__ bo, float* __restrict__ Out) {
  __shared__ __align__(16) bf16 Bt[64 * 32];   // [n][k] transposed Wo tile

  const int tid  = threadIdx.x;
  const int lane = tid & 31;
  const int wave = tid >> 5;
  const int lrow = lane & 15;
  const int hi   = (lane >> 4) & 1;

  const int m0 = blockIdx.x * 64;
  const int n0 = blockIdx.y * 64;
  const int msub = (wave & 3) * 16;
  const int nsub = (wave >> 2) * 32;

  v8f acc0 = {}, acc1 = {};
  const bf16* arow = Hd + (size_t)(m0 + msub + lrow) * NE;

  for (int e0 = 0; e0 < NE; e0 += 32) {
    // Bt[n][k] = Wo[e0+k][n0+n], packed pairs along k
    #pragma unroll
    for (int i = 0; i < 4; ++i) {
      int p = tid + i * 256;
      int n = p >> 4, kp = (p & 15) * 2;
      float w0 = Wo[(size_t)(e0 + kp) * NE + n0 + n];
      float w1 = Wo[(size_t)(e0 + kp + 1) * NE + n0 + n];
      ((unsigned*)Bt)[n * 16 + (p & 15)] = pack2(w0, w1);
    }
    __syncthreads();

    v16bf a  = frag_a(arow + e0, hi);
    v16bf b0 = frag_b(&Bt[(nsub + lrow) * 32], hi);
    v16bf b1 = frag_b(&Bt[(nsub + 16 + lrow) * 32], hi);
    acc0 = wmma_bf16(a, b0, acc0);
    acc1 = wmma_bf16(a, b1, acc1);
    __syncthreads();
  }

  #pragma unroll
  for (int r = 0; r < 8; ++r) {
    const int row = m0 + msub + hi * 8 + r;
    const int c0  = n0 + nsub + lrow;
    const int c1  = c0 + 16;
    Out[(size_t)row * NE + c0] = acc0[r] + bo[c0];
    Out[(size_t)row * NE + c1] = acc1[r] + bo[c1];
  }
}

// ---------------------------------------------------------------------------
// kernel_launch: inputs in setup_inputs() order.
//   0 queries [B,S,E] f32 | 1 keys | 2 values | 3 attn_mask (unused, causal
//   handled analytically) | 4 Wq [H,E,D] | 5 bq | 6 Wk | 7 bk | 8 Wv | 9 bv
//   | 10 Wo [H*D,E] | 11 bo
// Workspace: Qp,Kp,Vp [B,H,S,D] bf16 + heads [B,S,H*D] bf16  (~48 MB).
// ---------------------------------------------------------------------------
extern "C" void kernel_launch(void* const* d_in, const int* in_sizes, int n_in,
                              void* d_out, int out_size, void* d_ws, size_t ws_size,
                              hipStream_t stream) {
  const float* queries = (const float*)d_in[0];
  const float* keys    = (const float*)d_in[1];
  const float* values  = (const float*)d_in[2];
  const float* Wq = (const float*)d_in[4];
  const float* bq = (const float*)d_in[5];
  const float* Wk = (const float*)d_in[6];
  const float* bk = (const float*)d_in[7];
  const float* Wv = (const float*)d_in[8];
  const float* bv = (const float*)d_in[9];
  const float* Wo = (const float*)d_in[10];
  const float* bo = (const float*)d_in[11];
  float* out = (float*)d_out;

  const size_t nQKV = (size_t)NB * NH * NS * ND;   // 6,291,456 elements
  bf16* Qp = (bf16*)d_ws;
  bf16* Kp = Qp + nQKV;
  bf16* Vp = Kp + nQKV;
  bf16* Hd = Vp + nQKV;                            // [B,S,H*D]

  const dim3 blk(256);

  const dim3 gProj(NS / 64, NB * NH);
  qkv_proj_kernel<<<gProj, blk, 0, stream>>>(queries, Wq, bq, Qp);
  qkv_proj_kernel<<<gProj, blk, 0, stream>>>(keys,    Wk, bk, Kp);
  qkv_proj_kernel<<<gProj, blk, 0, stream>>>(values,  Wv, bv, Vp);

  const dim3 gAttn(NS / 128, NB * NH);
  attn_kernel<<<gAttn, blk, 0, stream>>>(Qp, Kp, Vp, Hd);

  const dim3 gOut((NB * NS) / 64, NE / 64);
  out_proj_kernel<<<gOut, blk, 0, stream>>>(Hd, Wo, bo, out);
}
